// ParaGraph_58445914964481
// MI455X (gfx1250) — matrix operand order repaired and checked by reading, not compiled, not run on hardware
//
#include <hip/hip_runtime.h>

#define NN 100000
#define EE 600000
#define DD 128

typedef __bf16 bf16;
typedef __attribute__((ext_vector_type(16))) __bf16 v16bf;
typedef __attribute__((ext_vector_type(8)))  __bf16 v8bf;
typedef __attribute__((ext_vector_type(8)))  float  v8f;

// ---------------------------------------------------------------------------
// small elementwise / setup kernels
// ---------------------------------------------------------------------------

// feature projection: out[n,d] = sum_k x[n,k] * W[d,k]  (K=16), store bf16
__global__ void proj_kernel(const float* __restrict__ x, const float* __restrict__ W,
                            bf16* __restrict__ out) {
    int idx = blockIdx.x * blockDim.x + threadIdx.x;
    if (idx >= NN * DD) return;
    int n = idx >> 7, d = idx & 127;
    const float* xr = x + n * 16;
    const float* wr = W + d * 16;
    float s = 0.f;
#pragma unroll
    for (int k = 0; k < 16; ++k) s += xr[k] * wr[k];
    out[idx] = (bf16)s;
}

__global__ void cvt_f32_bf16(const float* __restrict__ in, bf16* __restrict__ out, int n) {
    int i = blockIdx.x * blockDim.x + threadIdx.x;
    if (i < n) out[i] = (bf16)in[i];
}

// v[k] = sum_d a[d] * W[d*128 + k]   (one block of 128 threads)
__global__ void attvec_kernel(const float* __restrict__ a, const float* __restrict__ W,
                              float* __restrict__ v) {
    int k = threadIdx.x;
    float s = 0.f;
    for (int d = 0; d < DD; ++d) s += a[d] * W[d * DD + k];
    v[k] = s;
}

// logits[n] = feat[n,:] . v
__global__ void logits_kernel(const bf16* __restrict__ feat, const float* __restrict__ v,
                              float* __restrict__ out) {
    int n = blockIdx.x * blockDim.x + threadIdx.x;
    if (n >= NN) return;
    const bf16* f = feat + (size_t)n * DD;
    float s = 0.f;
    for (int k = 0; k < DD; ++k) s += (float)f[k] * v[k];
    out[n] = s;
}

__global__ void zero_kernel(float* __restrict__ p, int n) {
    int i = blockIdx.x * blockDim.x + threadIdx.x;
    if (i < n) p[i] = 0.f;
}

__global__ void init_md_kernel(float* __restrict__ m, float* __restrict__ denom) {
    int n = blockIdx.x * blockDim.x + threadIdx.x;
    if (n < NN) { m[n] = -1e30f; denom[n] = 0.f; }
}

// ob[n,d] = bf16(out[n,d] + bias[d])
__global__ void addbias_cvt_kernel(const float* __restrict__ out, const float* __restrict__ bias,
                                   bf16* __restrict__ ob) {
    int i = blockIdx.x * blockDim.x + threadIdx.x;
    if (i >= NN * DD) return;
    ob[i] = (bf16)(out[i] + bias[i & 127]);
}

// ---------------------------------------------------------------------------
// edge kernels (segment softmax + scatter)
// ---------------------------------------------------------------------------

__device__ __forceinline__ float leaky02(float x) { return x > 0.f ? x : 0.2f * x; }

__device__ __forceinline__ void atomicMaxF(float* addr, float val) {
    int* ai = reinterpret_cast<int*>(addr);
    int cur = __float_as_int(*addr);
    while (__int_as_float(cur) < val) {
        int old = atomicCAS(ai, cur, __float_as_int(val));
        if (old == cur) break;
        cur = old;
    }
}

__global__ void edge_max_kernel(const int* __restrict__ src, const int* __restrict__ dst,
                                const float* __restrict__ el, const float* __restrict__ er,
                                float* __restrict__ m) {
    int e = blockIdx.x * blockDim.x + threadIdx.x;
    if (e >= EE) return;
    float x = leaky02(el[src[e]] + er[dst[e]]);
    atomicMaxF(&m[dst[e]], x);
}

__global__ void edge_sum_kernel(const int* __restrict__ src, const int* __restrict__ dst,
                                const float* __restrict__ el, const float* __restrict__ er,
                                const float* __restrict__ m, float* __restrict__ denom) {
    int e = blockIdx.x * blockDim.x + threadIdx.x;
    if (e >= EE) return;
    int t = dst[e];
    float x = leaky02(el[src[e]] + er[t]);
    atomicAdd(&denom[t], expf(x - m[t]));
}

// block = 256 threads = 2 edges x 128 feature lanes
__global__ void edge_scatter_kernel(const int* __restrict__ src, const int* __restrict__ dst,
                                    const float* __restrict__ el, const float* __restrict__ er,
                                    const float* __restrict__ m, const float* __restrict__ denom,
                                    const float* __restrict__ hsrc, float* __restrict__ out) {
    int e = blockIdx.x * 2 + (threadIdx.x >> 7);
    if (e >= EE) return;
    int d = threadIdx.x & 127;
    int s = src[e], t = dst[e];
    float x = leaky02(el[s] + er[t]);
    float alpha = expf(x - m[t]) / denom[t];
    atomicAdd(&out[(size_t)t * DD + d], hsrc[(size_t)s * DD + d] * alpha);
}

// ---------------------------------------------------------------------------
// WMMA bf16 GEMM:  C[M x 128] = relu?( A1 @ B[:, :128]^T  (+ A2 @ B[:, 128:256]^T) )
//   A1, A2 row-major [M x 128] bf16;  B row-major [128 x ldb] bf16 (row = out col d)
//   block = 256 threads = 8 waves; wave w -> output cols [16w, 16w+16); block -> 16 rows
// ---------------------------------------------------------------------------

__device__ __forceinline__ v16bf load_fragA(const bf16* p, int hi) {
    // A 16x32 bf16 layout: lane(lo16) rows; elems 0..7 = K(8*hi ..), elems 8..15 = K(16+8*hi ..)
    v8bf lo = *(const v8bf*)(p + 8 * hi);
    v8bf hh = *(const v8bf*)(p + 16 + 8 * hi);
    return __builtin_shufflevector(lo, hh, 0, 1, 2, 3, 4, 5, 6, 7, 8, 9, 10, 11, 12, 13, 14, 15);
}

__device__ __forceinline__ v16bf load_fragB(const bf16* p, int hi) {
    // B 32x16 bf16 layout: lane = column; elems 0..7 = K(8*hi ..), elems 8..15 = K(16+8*hi ..)
    v8bf lo = *(const v8bf*)(p + 8 * hi);
    v8bf hh = *(const v8bf*)(p + 16 + 8 * hi);
    return __builtin_shufflevector(lo, hh, 0, 1, 2, 3, 4, 5, 6, 7, 8, 9, 10, 11, 12, 13, 14, 15);
}

__global__ __launch_bounds__(256)
void gemm_bf16_kernel(const bf16* __restrict__ A1, const bf16* __restrict__ A2,
                      const bf16* __restrict__ B, int ldb,
                      float* __restrict__ C, int M, int doRelu) {
    const int lane  = threadIdx.x & 31;
    const int wave  = threadIdx.x >> 5;       // 0..7 -> 16-col tile of the 128 outputs
    const int mBase = blockIdx.x << 4;
    if (mBase >= M) return;                    // uniform per block
    const int nBase = wave << 4;
    const int r     = lane & 15;
    const int hi    = lane >> 4;

    const bf16* aP = A1 + (size_t)(mBase + r) * DD;
    const bf16* bP = B + (size_t)(nBase + r) * ldb;

    v8f acc = {};
#pragma unroll
    for (int kc = 0; kc < DD; kc += 32) {
        v16bf a = load_fragA(aP + kc, hi);
        v16bf b = load_fragB(bP + kc, hi);
        acc = __builtin_amdgcn_wmma_f32_16x16x32_bf16(false, a, false, b, (short)0, acc,
                                                      false, false);
    }
    if (A2) {
        const bf16* a2P = A2 + (size_t)(mBase + r) * DD;
#pragma unroll
        for (int kc = 0; kc < DD; kc += 32) {
            v16bf a = load_fragA(a2P + kc, hi);
            v16bf b = load_fragB(bP + DD + kc, hi);
            acc = __builtin_amdgcn_wmma_f32_16x16x32_bf16(false, a, false, b, (short)0, acc,
                                                          false, false);
        }
    }
    // C/D layout: VGPR i, lanes 0-15: (M=i, N=lane); lanes 16-31: (M=8+i, N=lane-16)
#pragma unroll
    for (int i = 0; i < 8; ++i) {
        float v = acc[i];
        if (doRelu) v = fmaxf(v, 0.f);
        C[(size_t)(mBase + i + 8 * hi) * DD + nBase + r] = v;
    }
}

// ---------------------------------------------------------------------------
// launch
// ---------------------------------------------------------------------------

extern "C" void kernel_launch(void* const* d_in, const int* in_sizes, int n_in,
                              void* d_out, int out_size, void* d_ws, size_t ws_size,
                              hipStream_t stream) {
    const float* x_net   = (const float*)d_in[0];
    const float* x_cell  = (const float*)d_in[1];
    const float* Wp_net  = (const float*)d_in[2];
    const float* Wp_cell = (const float*)d_in[3];
    const float* Wg[3]   = {(const float*)d_in[4], (const float*)d_in[7], (const float*)d_in[10]};
    const float* al[3]   = {(const float*)d_in[5], (const float*)d_in[8], (const float*)d_in[11]};
    const float* ar[3]   = {(const float*)d_in[6], (const float*)d_in[9], (const float*)d_in[12]};
    const float* Wl      = (const float*)d_in[13];
    const float* bias    = (const float*)d_in[14];
    const int* srcs[3]   = {(const int*)d_in[15], (const int*)d_in[17], (const int*)d_in[19]};
    const int* dsts[3]   = {(const int*)d_in[16], (const int*)d_in[18], (const int*)d_in[20]};

    float* out_net  = (float*)d_out;              // [N, D]
    float* out_cell = out_net + (size_t)NN * DD;  // [N, D]
    float* outP[2]  = {out_net, out_cell};        // 0 = net, 1 = cell

    // ---- carve workspace (~130 MB) ----
    char* w = (char*)d_ws;
    auto carve = [&](size_t bytes) -> char* {
        char* p = w;
        w += (bytes + 255) & ~(size_t)255;
        return p;
    };
    bf16* feat_bf[2];
    feat_bf[0]  = (bf16*)carve((size_t)NN * DD * sizeof(bf16));   // net
    feat_bf[1]  = (bf16*)carve((size_t)NN * DD * sizeof(bf16));   // cell
    float* hsrc = (float*)carve((size_t)NN * DD * sizeof(float));
    bf16* outb  = (bf16*)carve((size_t)NN * DD * sizeof(bf16));
    bf16* Wg_bf[3];
    for (int i = 0; i < 3; ++i) Wg_bf[i] = (bf16*)carve((size_t)DD * DD * sizeof(bf16));
    bf16* Wl_bf = (bf16*)carve((size_t)DD * 2 * DD * sizeof(bf16));
    float* vl[3];
    float* vr[3];
    for (int i = 0; i < 3; ++i) {
        vl[i] = (float*)carve(DD * sizeof(float));
        vr[i] = (float*)carve(DD * sizeof(float));
    }
    float* el    = (float*)carve((size_t)NN * sizeof(float));
    float* er    = (float*)carve((size_t)NN * sizeof(float));
    float* m     = (float*)carve((size_t)NN * sizeof(float));
    float* denom = (float*)carve((size_t)NN * sizeof(float));

    const int gND    = (NN * DD + 255) / 256;  // 50000
    const int gN     = (NN + 255) / 256;
    const int gE     = (EE + 255) / 256;
    const int gGemm  = NN / 16;                // 6250 (N divisible by 16)

    // ---- one-time (per launch) prep ----
    proj_kernel<<<gND, 256, 0, stream>>>(x_net, Wp_net, feat_bf[0]);
    proj_kernel<<<gND, 256, 0, stream>>>(x_cell, Wp_cell, feat_bf[1]);
    for (int i = 0; i < 3; ++i)
        cvt_f32_bf16<<<(DD * DD + 255) / 256, 256, 0, stream>>>(Wg[i], Wg_bf[i], DD * DD);
    cvt_f32_bf16<<<(DD * 2 * DD + 255) / 256, 256, 0, stream>>>(Wl, Wl_bf, DD * 2 * DD);
    for (int i = 0; i < 3; ++i) {
        attvec_kernel<<<1, 128, 0, stream>>>(al[i], Wg[i], vl[i]);
        attvec_kernel<<<1, 128, 0, stream>>>(ar[i], Wg[i], vr[i]);
    }
    zero_kernel<<<(2 * NN * DD + 255) / 256, 256, 0, stream>>>(out_net, 2 * NN * DD);

    // relation r: src-type, dst-type  (0=net, 1=cell)
    const int srcT[3] = {1, 0, 0};  // cell->net, net->cell, net->net
    const int dstT[3] = {0, 1, 0};

    for (int r = 0; r < 3; ++r) {
        // attention logits per node (el for src type, er for dst type)
        logits_kernel<<<gN, 256, 0, stream>>>(feat_bf[srcT[r]], vl[r], el);
        logits_kernel<<<gN, 256, 0, stream>>>(feat_bf[dstT[r]], vr[r], er);
        // h_src = feat_src @ Wg^T   (WMMA bf16, f32 accumulate)
        gemm_bf16_kernel<<<gGemm, 256, 0, stream>>>(feat_bf[srcT[r]], nullptr, Wg_bf[r], DD,
                                                    hsrc, NN, 0);
        // segment softmax over incoming edges of each dst
        init_md_kernel<<<gN, 256, 0, stream>>>(m, denom);
        edge_max_kernel<<<gE, 256, 0, stream>>>(srcs[r], dsts[r], el, er, m);
        edge_sum_kernel<<<gE, 256, 0, stream>>>(srcs[r], dsts[r], el, er, m, denom);
        // out[dst] += alpha * h_src[src]
        edge_scatter_kernel<<<EE / 2, 256, 0, stream>>>(srcs[r], dsts[r], el, er, m, denom,
                                                        hsrc, outP[dstT[r]]);
        // fused concat+linear+relu for BOTH node types (faithful to reference)
        for (int nt = 0; nt < 2; ++nt) {
            addbias_cvt_kernel<<<gND, 256, 0, stream>>>(outP[nt], bias, outb);
            gemm_bf16_kernel<<<gGemm, 256, 0, stream>>>(feat_bf[nt], outb, Wl_bf, 2 * DD,
                                                        outP[nt], NN, 1);
        }
    }
}
